// dilated_dgcnn_45251775430978
// MI455X (gfx1250) — matrix operand order repaired and checked by reading, not compiled, not run on hardware
//
#include <hip/hip_runtime.h>
#include <hip/hip_bf16.h>

#define Npts 12000
#define Knbr 20
#define DKc  40
#define NT   9

typedef __attribute__((ext_vector_type(16))) _Float16     v16h;
typedef __attribute__((ext_vector_type(8)))  _Float16     v8h;
typedef __attribute__((ext_vector_type(8)))  float        v8f;
typedef __attribute__((ext_vector_type(4)))  unsigned int u32x4;
typedef __attribute__((ext_vector_type(8)))  int          i32x8;
typedef __attribute__((ext_vector_type(4)))  int          i32x4;

// ---------------------------------------------------------------------------
// Build initial feature x0[12][n] = [pos.T ; one_hot(node_type)]
// ---------------------------------------------------------------------------
__global__ void build_x0_kernel(const float* __restrict__ pos, const int* __restrict__ nt,
                                float* __restrict__ x0, int n) {
    int i = blockIdx.x * blockDim.x + threadIdx.x;
    if (i >= n) return;
    x0[0 * n + i] = pos[3 * i + 0];
    x0[1 * n + i] = pos[3 * i + 1];
    x0[2 * n + i] = pos[3 * i + 2];
    int t = nt[i];
    for (int c = 0; c < NT; ++c) x0[(3 + c) * n + i] = (c == t) ? 1.f : 0.f;
}

// ---------------------------------------------------------------------------
// Weight pre-convert: Wf[oc][k] = f16(W[oc][k]) for k<IC else 0   (ICpad cols)
// ---------------------------------------------------------------------------
__global__ void wcvt_kernel(const float* __restrict__ W, _Float16* __restrict__ Wf,
                            int OC, int IC, int ICpad) {
    int e = blockIdx.x * blockDim.x + threadIdx.x;
    if (e >= OC * ICpad) return;
    int oc = e / ICpad, k = e - oc * ICpad;
    Wf[e] = (_Float16)(k < IC ? W[(size_t)oc * IC + k] : 0.f);
}

// ---------------------------------------------------------------------------
// kNN top-40 (ascending, self first) fused with FPS 40 -> 20.
// One thread per point; per-thread scratch arrays live in LDS.
// ---------------------------------------------------------------------------
#define KNN_BLK 32
__global__ __launch_bounds__(KNN_BLK)
void knn_fps_kernel(const float* __restrict__ pos, int* __restrict__ idx_out, int n) {
    __shared__ float s_d [DKc * KNN_BLK];
    __shared__ int   s_i [DKc * KNN_BLK];
    __shared__ float s_px[DKc * KNN_BLK];
    __shared__ float s_py[DKc * KNN_BLK];
    __shared__ float s_pz[DKc * KNN_BLK];
    __shared__ float s_m [DKc * KNN_BLK];
    int t = threadIdx.x;
    int i = blockIdx.x * KNN_BLK + t;
    if (i >= n) return;
    float px = pos[3 * i], py = pos[3 * i + 1], pz = pos[3 * i + 2];
    for (int s = 0; s < DKc; ++s) { s_d[s * KNN_BLK + t] = 3.4e38f; s_i[s * KNN_BLK + t] = 0; }
    float cmax = 3.4e38f; int cslot = 0;
    for (int j = 0; j < n; ++j) {
        float dx = pos[3 * j] - px, dy = pos[3 * j + 1] - py, dz = pos[3 * j + 2] - pz;
        float d = dx * dx + dy * dy + dz * dz;
        if (d < cmax) {
            s_d[cslot * KNN_BLK + t] = d;
            s_i[cslot * KNN_BLK + t] = j;
            cmax = -1.f;
            for (int s = 0; s < DKc; ++s) {
                float v = s_d[s * KNN_BLK + t];
                if (v > cmax) { cmax = v; cslot = s; }
            }
        }
    }
    for (int a = 0; a < DKc - 1; ++a) {               // sort ascending (top_k is sorted)
        int bm = a; float vm = s_d[a * KNN_BLK + t];
        for (int b2 = a + 1; b2 < DKc; ++b2) {
            float v = s_d[b2 * KNN_BLK + t];
            if (v < vm) { vm = v; bm = b2; }
        }
        if (bm != a) {
            float td = s_d[a * KNN_BLK + t];
            s_d[a * KNN_BLK + t] = s_d[bm * KNN_BLK + t]; s_d[bm * KNN_BLK + t] = td;
            int ti = s_i[a * KNN_BLK + t];
            s_i[a * KNN_BLK + t] = s_i[bm * KNN_BLK + t]; s_i[bm * KNN_BLK + t] = ti;
        }
    }
    float p0x = 0, p0y = 0, p0z = 0;
    for (int s = 0; s < DKc; ++s) {
        int j = s_i[s * KNN_BLK + t];
        float qx = pos[3 * j], qy = pos[3 * j + 1], qz = pos[3 * j + 2];
        s_px[s * KNN_BLK + t] = qx; s_py[s * KNN_BLK + t] = qy; s_pz[s * KNN_BLK + t] = qz;
        if (s == 0) { p0x = qx; p0y = qy; p0z = qz; }
    }
    for (int s = 0; s < DKc; ++s) {
        float dx = s_px[s * KNN_BLK + t] - p0x, dy = s_py[s * KNN_BLK + t] - p0y, dz = s_pz[s * KNN_BLK + t] - p0z;
        s_m[s * KNN_BLK + t] = dx * dx + dy * dy + dz * dz;
    }
    idx_out[i * Knbr + 0] = s_i[0 * KNN_BLK + t];
    for (int sel = 1; sel < Knbr; ++sel) {
        float best = -1.f; int bs = 0;
        for (int s = 0; s < DKc; ++s) {
            float v = s_m[s * KNN_BLK + t];
            if (v > best) { best = v; bs = s; }   // strict > == argmax first-occurrence
        }
        idx_out[i * Knbr + sel] = s_i[bs * KNN_BLK + t];
        float qx = s_px[bs * KNN_BLK + t], qy = s_py[bs * KNN_BLK + t], qz = s_pz[bs * KNN_BLK + t];
        for (int s = 0; s < DKc; ++s) {
            float dx = s_px[s * KNN_BLK + t] - qx, dy = s_py[s * KNN_BLK + t] - qy, dz = s_pz[s * KNN_BLK + t] - qz;
            float d = dx * dx + dy * dy + dz * dz;
            float cur = s_m[s * KNN_BLK + t];
            s_m[s * KNN_BLK + t] = d < cur ? d : cur;
        }
    }
}

// ---------------------------------------------------------------------------
// Graph feature gather -> f16 B operand, TRANSPOSED [M][ICpad] (K-contiguous):
//   Bt[col][0:C]=nbr-ctr, Bt[col][C:2C]=ctr, Bt[col][2C:ICpad]=0
// ---------------------------------------------------------------------------
__global__ void gather_kernel(const float* __restrict__ feat, int C, const int* __restrict__ idx,
                              _Float16* __restrict__ Bt, int ICpad, int n) {
    int col = blockIdx.x * blockDim.x + threadIdx.x;
    int M = n * Knbr;
    if (col >= M) return;
    int i = col / Knbr;
    int j = idx[col];
    _Float16* row = Bt + (size_t)col * ICpad;
    for (int c = 0; c < C; ++c) {
        float ctr = feat[(size_t)c * n + i];
        float nb  = feat[(size_t)c * n + j];
        row[c]     = (_Float16)(nb - ctr);
        row[C + c] = (_Float16)ctr;
    }
    for (int c = 2 * C; c < ICpad; ++c) row[c] = (_Float16)0.f;
}

// ---------------------------------------------------------------------------
// WMMA GEMM: Y[OC][M] = Wf16[OC][ICpad] * BmT[M][ICpad]^T   (f32 accumulate)
// A tile (16 x ICpad f16) is DMA'd into LDS once per block via the Tensor
// Data Mover; both A and B fragments are 2x b128 loads per k-step.
//   A 16x32 f16 : row = lane&15; K = 16*(lane>>4) + e
//   B 32x16 f16 : col = lane&15; K = 16*(lane>>4) + e   (K-contiguous in BmT)
//   C/D 16x16 f32: row = v + 8*(lane>>4); col = lane&15
// ---------------------------------------------------------------------------
__global__ __launch_bounds__(256)
void gemm_wmma_kernel(const _Float16* __restrict__ Wf, const _Float16* __restrict__ BmT,
                      float* __restrict__ Y, int ICpad, int M) {
    extern __shared__ _Float16 ldsW[];          // 16 * ICpad halves
    int oc0 = blockIdx.y << 4;

    if (threadIdx.x < 32) {                     // wave 0 issues the TDM load
        unsigned lds_off = (unsigned)(size_t)&ldsW[0];
        unsigned long long ga = (unsigned long long)(size_t)(Wf + (size_t)oc0 * ICpad);
        u32x4 g0;
        g0[0] = 1u;                                             // count=1
        g0[1] = lds_off;                                        // lds_addr
        g0[2] = (unsigned)ga;                                   // global_addr[31:0]
        g0[3] = (unsigned)((ga >> 32) & 0x01FFFFFFu) | 0x80000000u; // addr[56:32] | type=2
        i32x8 g1;
        g1[0] = 1 << 16;                                        // data_size = 2 bytes
        g1[1] = (ICpad & 0xFFFF) << 16;                         // tensor_dim0[15:0]
        g1[2] = ((ICpad >> 16) & 0xFFFF) | (16 << 16);          // td0[31:16] | tensor_dim1=16
        g1[3] = (ICpad << 16);                                  // td1[31:16]=0 | tile_dim0=ICpad
        g1[4] = 16;                                             // tile_dim1=16, tile_dim2=0
        g1[5] = ICpad;                                          // tensor_dim0_stride[31:0]
        g1[6] = 0;
        g1[7] = 0;
        i32x4 z4 = {0, 0, 0, 0};
        i32x8 z8 = {0, 0, 0, 0, 0, 0, 0, 0};
        __builtin_amdgcn_tensor_load_to_lds(g0, g1, z4, z4, z8, 0);
        __builtin_amdgcn_s_wait_tensorcnt(0);
    }
    __syncthreads();

    int lane = threadIdx.x & 31;
    int wave = threadIdx.x >> 5;
    int tilesM = M >> 4;
    int colTile = blockIdx.x * 8 + wave;
    if (colTile >= tilesM) return;              // whole-wave exit keeps EXEC all-1s
    int half = lane >> 4;
    int l16  = lane & 15;
    int col  = (colTile << 4) + l16;
    const _Float16* bbase = BmT + (size_t)col * ICpad + 16 * half;

    v8f acc = {};
#pragma unroll 2
    for (int kk = 0; kk < ICpad; kk += 32) {
        const v8h* apv = (const v8h*)(ldsW + (size_t)l16 * ICpad + kk + 16 * half);
        v8h a0 = apv[0];                        // K = kk + 16*half + 0..7
        v8h a1 = apv[1];                        // K = kk + 16*half + 8..15
        const v8h* bpv = (const v8h*)(bbase + kk);
        v8h b0 = bpv[0];                        // K = kk + 16*half + 0..7
        v8h b1 = bpv[1];                        // K = kk + 16*half + 8..15
        v16h a, b;
#pragma unroll
        for (int e = 0; e < 8; ++e) {
            a[e] = a0[e]; a[8 + e] = a1[e];
            b[e] = b0[e]; b[8 + e] = b1[e];
        }
        acc = __builtin_amdgcn_wmma_f32_16x16x32_f16(false, a, false, b,
                                                     (short)0, acc, false, false);
    }
#pragma unroll
    for (int v = 0; v < 8; ++v)
        Y[(size_t)(oc0 + v + 8 * half) * M + col] = acc[v];
}

// ---------------------------------------------------------------------------
// GroupNorm stats: one block per group, reduce over (C/G)*M elements
// ---------------------------------------------------------------------------
__global__ __launch_bounds__(256)
void gn_stats_kernel(const float* __restrict__ Y, int C, int G, int M, float* __restrict__ stats) {
    int grp = blockIdx.x;
    int cpg = C / G;
    size_t total = (size_t)cpg * (size_t)M;
    const float* base = Y + (size_t)grp * total;
    float s = 0.f, q = 0.f;
    for (size_t e = threadIdx.x; e < total; e += blockDim.x) {
        float v = base[e]; s += v; q += v * v;
    }
    __shared__ float sh_s[256], sh_q[256];
    sh_s[threadIdx.x] = s; sh_q[threadIdx.x] = q;
    __syncthreads();
    for (int o = 128; o > 0; o >>= 1) {
        if ((int)threadIdx.x < o) {
            sh_s[threadIdx.x] += sh_s[threadIdx.x + o];
            sh_q[threadIdx.x] += sh_q[threadIdx.x + o];
        }
        __syncthreads();
    }
    if (threadIdx.x == 0) {
        float inv  = 1.f / (float)total;
        float mean = sh_s[0] * inv;
        float var  = sh_q[0] * inv - mean * mean;
        var = var < 0.f ? 0.f : var;
        stats[2 * grp + 0] = mean;
        stats[2 * grp + 1] = rsqrtf(var + 1e-5f);
    }
}

// ---------------------------------------------------------------------------
// GroupNorm apply + LeakyReLU(0.2), elementwise in place (no shadow)
// ---------------------------------------------------------------------------
__global__ __launch_bounds__(256)
void gn_apply_kernel(float* __restrict__ Y, int C, int G, int M,
                     const float* __restrict__ stats,
                     const float* __restrict__ gamma, const float* __restrict__ beta) {
    size_t e = (size_t)blockIdx.x * blockDim.x + threadIdx.x;
    size_t tot = (size_t)C * (size_t)M;
    if (e >= tot) return;
    int c   = (int)(e / (size_t)M);
    int cpg = C / G;
    int grp = c / cpg;
    float mean = stats[2 * grp], rstd = stats[2 * grp + 1];
    float v = (Y[e] - mean) * rstd * gamma[c] + beta[c];
    Y[e] = v >= 0.f ? v : 0.2f * v;
}

// ---------------------------------------------------------------------------
// GroupNorm apply + LeakyReLU, column-parallel: f32 in place + f16 transposed
// shadow Bt[M][C] (K-contiguous B operand for the next GEMM)
// ---------------------------------------------------------------------------
__global__ __launch_bounds__(256)
void gn_apply_colT_kernel(float* __restrict__ Y, int C, int G, int M,
                          const float* __restrict__ stats,
                          const float* __restrict__ gamma, const float* __restrict__ beta,
                          _Float16* __restrict__ Bt) {
    int col = blockIdx.x * blockDim.x + threadIdx.x;
    if (col >= M) return;
    int cpg = C / G;
    _Float16* brow = Bt + (size_t)col * C;
    for (int c = 0; c < C; ++c) {
        int grp = c / cpg;
        float mean = stats[2 * grp], rstd = stats[2 * grp + 1];
        float v = (Y[(size_t)c * M + col] - mean) * rstd * gamma[c] + beta[c];
        v = v >= 0.f ? v : 0.2f * v;
        Y[(size_t)c * M + col] = v;
        brow[c] = (_Float16)v;
    }
}

// max over K neighbors: [C][n*K] -> f32 [C][n] + f16 transposed slice of
// xchT[n][192] at channel offset cofs
__global__ void maxk_kernel(const float* __restrict__ Y, float* __restrict__ out,
                            _Float16* __restrict__ xchT, int cofs, int C, int n) {
    int e = blockIdx.x * blockDim.x + threadIdx.x;
    if (e >= C * n) return;
    int c = e / n, i = e - c * n;
    size_t M = (size_t)n * Knbr;
    const float* row = Y + (size_t)c * M + (size_t)i * Knbr;
    float m = row[0];
    for (int k = 1; k < Knbr; ++k) m = row[k] > m ? row[k] : m;
    out[(size_t)c * n + i] = m;
    xchT[(size_t)i * 192 + cofs + c] = (_Float16)m;
}

// global max over points: [C][n] -> [C]
__global__ void maxn_kernel(const float* __restrict__ H, float* __restrict__ hmax, int C, int n) {
    int c = blockIdx.x * blockDim.x + threadIdx.x;
    if (c >= C) return;
    const float* row = H + (size_t)c * n;
    float m = row[0];
    for (int i = 1; i < n; ++i) m = row[i] > m ? row[i] : m;
    hmax[c] = m;
}

// h7inT[n][1216] f16: cols 0..1023 broadcast hmax, 1024..1215 = xchT row
__global__ void build_h7in_kernel(const float* __restrict__ hmax, const _Float16* __restrict__ xchT,
                                  _Float16* __restrict__ H, int n) {
    int col = blockIdx.x * blockDim.x + threadIdx.x;
    if (col >= n) return;
    _Float16* hrow = H + (size_t)col * 1216;
    for (int r = 0; r < 1024; ++r) hrow[r] = (_Float16)hmax[r];
    const _Float16* xrow = xchT + (size_t)col * 192;
    for (int r = 0; r < 192; ++r) hrow[1024 + r] = xrow[r];
}

// out[i][o] = sum_c w9[o][c] * h8[c][i]
__global__ void final_kernel(const float* __restrict__ H8, const float* __restrict__ w9,
                             float* __restrict__ out, int n) {
    int i = blockIdx.x * blockDim.x + threadIdx.x;
    if (i >= n) return;
    for (int o = 0; o < 3; ++o) {
        float acc = 0.f;
        for (int c = 0; c < 256; ++c) acc += w9[o * 256 + c] * H8[(size_t)c * n + i];
        out[i * 3 + o] = acc;
    }
}

// ---------------------------------------------------------------------------
// Host-side helpers
// ---------------------------------------------------------------------------
static void run_gemm(const _Float16* Wf, int ICpad, const _Float16* BT, float* Y,
                     int OC, int M, hipStream_t s) {
    int tilesM = M / 16;
    dim3 grid((tilesM + 7) / 8, OC / 16);
    size_t shmem = (size_t)16 * ICpad * sizeof(_Float16);
    gemm_wmma_kernel<<<grid, 256, shmem, s>>>(Wf, BT, Y, ICpad, M);
}

static void run_gn(float* Y, int C, int G, int M, const float* gamma, const float* beta,
                   float* stats, _Float16* shadowT, hipStream_t s) {
    gn_stats_kernel<<<G, 256, 0, s>>>(Y, C, G, M, stats);
    if (shadowT) {
        gn_apply_colT_kernel<<<(M + 255) / 256, 256, 0, s>>>(Y, C, G, M, stats, gamma, beta, shadowT);
    } else {
        size_t tot = (size_t)C * (size_t)M;
        gn_apply_kernel<<<(unsigned)((tot + 255) / 256), 256, 0, s>>>(Y, C, G, M, stats, gamma, beta);
    }
}

extern "C" void kernel_launch(void* const* d_in, const int* in_sizes, int n_in,
                              void* d_out, int out_size, void* d_ws, size_t ws_size,
                              hipStream_t stream) {
    const int n = Npts;
    const int M = n * Knbr;   // 240000

    const float* pos   = (const float*)d_in[0];
    const int*   ntype = (const int*)d_in[1];
    const float *w1a=(const float*)d_in[2],  *g1a=(const float*)d_in[3],  *b1a=(const float*)d_in[4];
    const float *w1b=(const float*)d_in[5],  *g1b=(const float*)d_in[6],  *b1b=(const float*)d_in[7];
    const float *w2a=(const float*)d_in[8],  *g2a=(const float*)d_in[9],  *b2a=(const float*)d_in[10];
    const float *w2b=(const float*)d_in[11], *g2b=(const float*)d_in[12], *b2b=(const float*)d_in[13];
    const float *w5 =(const float*)d_in[14], *g5 =(const float*)d_in[15], *b5 =(const float*)d_in[16];
    const float *w6 =(const float*)d_in[17], *g6 =(const float*)d_in[18], *b6 =(const float*)d_in[19];
    const float *w7 =(const float*)d_in[20], *g7 =(const float*)d_in[21], *b7 =(const float*)d_in[22];
    const float *w8 =(const float*)d_in[23], *g8 =(const float*)d_in[24], *b8 =(const float*)d_in[25];
    const float *w9 =(const float*)d_in[26];
    float* out = (float*)d_out;

    // ---- workspace layout (all chunks 16B aligned) ----
    char* wsb = (char*)d_ws;
    int*      idx   = (int*)wsb;                  wsb += (size_t)n * Knbr * 4;      // n*K int
    float*    x0    = (float*)wsb;                wsb += (size_t)12 * n * 4;        // 12 x n f32
    _Float16* Bg    = (_Float16*)wsb;             wsb += (size_t)M * 128 * 2;       // gatherT / h7inT f16
    _Float16* Bs    = (_Float16*)wsb;             wsb += (size_t)M * 64  * 2;       // gn shadowT f16
    float*    Yb    = (float*)wsb;                wsb += (size_t)64 * M * 4;        // conv out f32
    float*    xc    = (float*)wsb;                wsb += (size_t)192 * n * 4;       // x1|x2|x3 f32
    _Float16* xchT  = (_Float16*)wsb;             wsb += (size_t)n * 192 * 2;       // [n][192] f16
    float*    h6    = (float*)wsb;                wsb += (size_t)1024 * n * 4;      // h6 / h8 f32
    float*    hmax  = (float*)wsb;                wsb += 1024 * 4;
    float*    stats = (float*)wsb;                wsb += 64 * 4;
    _Float16* Wf    = (_Float16*)wsb;                                               // padded f16 weights
    _Float16 *Wf1a = Wf,              *Wf1b = Wf1a + 64 * 32,   *Wf2a = Wf1b + 64 * 64;
    _Float16 *Wf2b = Wf2a + 64 * 128, *Wf5  = Wf2b + 64 * 64,   *Wf6  = Wf5  + 64 * 128;
    _Float16 *Wf7  = Wf6 + 1024 * 192, *Wf8 = Wf7 + 512 * 1216;

    // ---- pre-convert weights to padded f16 ----
    struct WC { const float* w; _Float16* wf; int OC, IC, ICpad; } wc[8] = {
        {w1a, Wf1a,   64,   24,   32}, {w1b, Wf1b,  64,  64,  64},
        {w2a, Wf2a,   64,  128,  128}, {w2b, Wf2b,  64,  64,  64},
        {w5,  Wf5,    64,  128,  128}, {w6,  Wf6, 1024, 192, 192},
        {w7,  Wf7,   512, 1216, 1216}, {w8,  Wf8,  256, 512, 512}};
    for (int q = 0; q < 8; ++q) {
        int tot = wc[q].OC * wc[q].ICpad;
        wcvt_kernel<<<(tot + 255) / 256, 256, 0, stream>>>(wc[q].w, wc[q].wf,
                                                           wc[q].OC, wc[q].IC, wc[q].ICpad);
    }

    build_x0_kernel<<<(n + 255) / 256, 256, 0, stream>>>(pos, ntype, x0, n);
    knn_fps_kernel<<<(n + KNN_BLK - 1) / KNN_BLK, KNN_BLK, 0, stream>>>(pos, idx, n);

    // ---- Block 1: x -> x1 (2C=24 padded to 32) ----
    gather_kernel<<<(M + 255) / 256, 256, 0, stream>>>(x0, 12, idx, Bg, 32, n);
    run_gemm(Wf1a, 32, Bg, Yb, 64, M, stream);
    run_gn(Yb, 64, 8, M, g1a, b1a, stats, Bs, stream);
    run_gemm(Wf1b, 64, Bs, Yb, 64, M, stream);
    run_gn(Yb, 64, 8, M, g1b, b1b, stats, nullptr, stream);
    maxk_kernel<<<(64 * n + 255) / 256, 256, 0, stream>>>(Yb, xc, xchT, 0, 64, n);

    // ---- Block 2: x1 -> x2 ----
    gather_kernel<<<(M + 255) / 256, 256, 0, stream>>>(xc, 64, idx, Bg, 128, n);
    run_gemm(Wf2a, 128, Bg, Yb, 64, M, stream);
    run_gn(Yb, 64, 8, M, g2a, b2a, stats, Bs, stream);
    run_gemm(Wf2b, 64, Bs, Yb, 64, M, stream);
    run_gn(Yb, 64, 8, M, g2b, b2b, stats, nullptr, stream);
    maxk_kernel<<<(64 * n + 255) / 256, 256, 0, stream>>>(Yb, xc + (size_t)64 * n, xchT, 64, 64, n);

    // ---- Block 3: x2 -> x3 ----
    gather_kernel<<<(M + 255) / 256, 256, 0, stream>>>(xc + (size_t)64 * n, 64, idx, Bg, 128, n);
    run_gemm(Wf5, 128, Bg, Yb, 64, M, stream);
    run_gn(Yb, 64, 16, M, g5, b5, stats, nullptr, stream);
    maxk_kernel<<<(64 * n + 255) / 256, 256, 0, stream>>>(Yb, xc + (size_t)128 * n, xchT, 128, 64, n);

    // ---- conv6 on concat(x1,x2,x3) = xchT [n][192] f16 ----
    run_gemm(Wf6, 192, xchT, h6, 1024, n, stream);
    run_gn(h6, 1024, 32, n, g6, b6, stats, nullptr, stream);
    maxn_kernel<<<(1024 + 255) / 256, 256, 0, stream>>>(h6, hmax, 1024, n);

    // ---- head: h7inT [n][1216] f16 -> conv7 -> conv8 -> w9 ----
    build_h7in_kernel<<<(n + 255) / 256, 256, 0, stream>>>(hmax, xchT, Bg, n);
    run_gemm(Wf7, 1216, Bg, Yb, 512, n, stream);          // h7 f32 fits in Yb
    run_gn(Yb, 512, 16, n, g7, b7, stats, Bs, stream);    // shadowT [n][512] fits in Bs
    run_gemm(Wf8, 512, Bs, h6, 256, n, stream);           // h8 f32 reuses h6
    run_gn(h6, 256, 16, n, g8, b8, stats, nullptr, stream);
    final_kernel<<<(n + 255) / 256, 256, 0, stream>>>(h6, w9, out, n);
}
